// GNN_43516608643276
// MI455X (gfx1250) — matrix operand order repaired and checked by reading, not compiled, not run on hardware
//
#include <hip/hip_runtime.h>

// ---------------------------------------------------------------------------
// GNN neighbor-attention, fused single-pass kernel for MI455X (gfx1250).
//
// Roofline: x_nb = 819 MB dominates -> ~37 us floor at 23.3 TB/s, provided
// x_nb is read from HBM exactly once. Main GEMM (52 GFLOP) runs on
// V_WMMA_F32_16X16X32_BF16 with the bf16 hi/lo split (3 WMMAs per tile step,
// ~fp32 accuracy) -- comfortably above the memory bound, unlike the K=4 f32
// WMMA which would be ~4x too slow. Streaming uses the CDNA5 async-to-LDS
// path (GLOBAL_LOAD_ASYNC_TO_LDS_B128, ASYNCcnt) to double-buffer the next
// node's tile behind the current node's WMMA work. Wn B-fragments are held
// in VGPRs for the entire node loop (loop-invariant).
// ---------------------------------------------------------------------------

#define DD   128   // hidden size
#define KNB  32    // neighbors per node
#define NPB  16    // nodes per block
#define NTHREADS 256  // 8 wave32s

typedef __attribute__((ext_vector_type(16))) __bf16 v16bf;
typedef __attribute__((ext_vector_type(8)))  __bf16 v8bf;
typedef __attribute__((ext_vector_type(8)))  float  v8f;

// bytes of dynamic LDS
#define SMEM_BYTES ((size_t)( 2*KNB*DD*4 /* f32 staging x2 */ \
                   + (2*DD*DD + 2*KNB*DD + 2*NPB*DD) * 2 /* bf16 */ \
                   + (NPB*DD + NPB*KNB + DD + DD + DD + KNB + KNB + DD) * 4 ))

__device__ __forceinline__ v16bf ld_frag(const __bf16* p0, const __bf16* p1) {
    v8bf a = *(const v8bf*)p0;
    v8bf b = *(const v8bf*)p1;
    return __builtin_shufflevector(a, b, 0,1,2,3,4,5,6,7,8,9,10,11,12,13,14,15);
}

// acc += (Ah+Al) x (Bh+Bl), dropping the lo*lo term (bf16x3 fp32 emulation)
__device__ __forceinline__ v8f wmma3(v16bf ah, v16bf al, v16bf bh, v16bf bl, v8f acc) {
    acc = __builtin_amdgcn_wmma_f32_16x16x32_bf16(false, ah, false, bh, (short)0, acc, false, false);
    acc = __builtin_amdgcn_wmma_f32_16x16x32_bf16(false, ah, false, bl, (short)0, acc, false, false);
    acc = __builtin_amdgcn_wmma_f32_16x16x32_bf16(false, al, false, bh, (short)0, acc, false, false);
    return acc;
}

// DMA one 16 KB tile (32x128 f32) global -> LDS via the async unit (ASYNCcnt).
// Generic shared pointers carry the LDS byte offset in their low 32 bits.
__device__ __forceinline__ void async_tile_16k(const float* __restrict__ g,
                                               float* stg, int tid) {
    uint32_t base = (uint32_t)(uintptr_t)stg;
    #pragma unroll
    for (int i = 0; i < (KNB*DD*4) / 16 / NTHREADS; ++i) {   // 4 x b128 per thread
        int idx = tid + i*NTHREADS;
        uint32_t loff = base + (uint32_t)idx * 16u;
        const float* gp = g + idx * 4;
        asm volatile("global_load_async_to_lds_b128 %0, %1, off"
                     :: "v"(loff), "v"(gp) : "memory");
    }
}

__global__ __launch_bounds__(NTHREADS)
void gnn_attn_kernel(const float* __restrict__ x,
                     const float* __restrict__ x_nb,
                     const float* __restrict__ weight,
                     const float* __restrict__ Wx,
                     const float* __restrict__ bx,
                     const float* __restrict__ Wn,
                     const float* __restrict__ bn,
                     const float* __restrict__ Ww,
                     const float* __restrict__ bw,
                     const float* __restrict__ Wl,
                     const float* __restrict__ bl,
                     float* __restrict__ out,
                     int Ntot)
{
    extern __shared__ __align__(16) char smem_raw[];
    float*  stg0 = (float*)smem_raw;         // [32*128] f32 staging (double buffered)
    float*  stg1 = stg0 + KNB*DD;
    __bf16* Whi = (__bf16*)(stg1 + KNB*DD);  // [128*128] weight hi (Wx, then Wn)
    __bf16* Wlo = Whi + DD*DD;               // [128*128] weight lo
    __bf16* nbh = Wlo + DD*DD;               // [32*128]  x_nb tile hi
    __bf16* nbl = nbh + KNB*DD;              // [32*128]  x_nb tile lo
    __bf16* xah = nbl + KNB*DD;              // [16*128]  x tile hi
    __bf16* xal = xah + NPB*DD;              // [16*128]  x tile lo
    float*  xsA = (float*)(xal + NPB*DD);    // [16*128]  x_s + (bx+bn+bw)
    float*  wgt = xsA + NPB*DD;              // [16*32]   edge weights
    float*  vWw = wgt + NPB*KNB;             // [128]
    float*  vWl = vWw + DD;                  // [128]
    float*  b3  = vWl + DD;                  // [128]     bx+bn+bw
    float*  sco = b3  + DD;                  // [32]      scores
    float*  att = sco + KNB;                 // [32]      softmax weights
    float*  osum= att + KNB;                 // [128]     output half-sum

    const int tid  = threadIdx.x;
    const int lane = tid & 31;
    const int wave = tid >> 5;
    const int l16  = lane & 15;
    const int lh   = lane >> 4;
    const long node0 = (long)blockIdx.x * NPB;

    // Kick off the async DMA of node 0's tile immediately: it overlaps the
    // entire weight-staging + x_s phase below.
    {
        long n0c = node0; if (n0c >= Ntot) n0c = Ntot - 1;
        async_tile_16k(x_nb + n0c * (long)(KNB*DD), stg0, tid);
    }

    // ---- Phase 0: stage Wx (split bf16), x-tile, edge weights, small vectors
    for (int i = tid; i < DD*DD; i += NTHREADS) {
        float f = Wx[i];
        __bf16 h = (__bf16)f;
        Whi[i] = h; Wlo[i] = (__bf16)(f - (float)h);
    }
    if (tid < DD) {
        vWw[tid] = Ww[tid];
        vWl[tid] = Wl[tid];
        b3[tid]  = bx[tid] + bn[tid] + bw[tid];
    }
    {
        const long lim = (long)Ntot * DD;
        for (int i = tid; i < NPB*DD; i += NTHREADS) {
            long gi = node0 * DD + i; if (gi >= lim) gi = lim - 1;
            float f = x[gi];
            __bf16 h = (__bf16)f;
            xah[i] = h; xal[i] = (__bf16)(f - (float)h);
        }
        const long limw = (long)Ntot * KNB;
        for (int i = tid; i < NPB*KNB; i += NTHREADS) {
            long gi = node0 * KNB + i; if (gi >= limw) gi = limw - 1;
            wgt[i] = weight[gi];
        }
    }
    __syncthreads();

    // ---- Phase A: x_s = x @ Wx^T (+fused biases) for the 16 nodes.
    //      One 16x16 N-tile per wave, K=128 in 4 steps of 32, bf16x3.
    {
        const int e = wave*16 + l16;         // output feature column
        v8f acc = {};
        #pragma unroll
        for (int ks = 0; ks < 4; ++ks) {
            const int ka = ks*32 + lh*8;
            v16bf ahf = ld_frag(xah + l16*DD + ka, xah + l16*DD + ka + 16);
            v16bf alf = ld_frag(xal + l16*DD + ka, xal + l16*DD + ka + 16);
            const int kb = ks*32 + lh*8;
            v16bf bhf = ld_frag(Whi + e*DD + kb, Whi + e*DD + kb + 16);
            v16bf blf = ld_frag(Wlo + e*DD + kb, Wlo + e*DD + kb + 16);
            acc = wmma3(ahf, alf, bhf, blf, acc);
        }
        const float bias = b3[e];
        #pragma unroll
        for (int r = 0; r < 8; ++r)          // C layout: VGPR r -> row r (+8 upper half)
            xsA[(r + 8*lh)*DD + e] = acc[r] + bias;
    }
    __syncthreads();

    // ---- Phase 1: replace LDS weights with Wn (split bf16)
    for (int i = tid; i < DD*DD; i += NTHREADS) {
        float f = Wn[i];
        __bf16 h = (__bf16)f;
        Whi[i] = h; Wlo[i] = (__bf16)(f - (float)h);
    }
    __syncthreads();

    const float bl0 = bl[0];

    // Each wave owns one 16-wide e-slice for the whole node loop; its Wn
    // B-fragments are loop-invariant -> keep them in VGPRs (8 x v16bf).
    const int eW = wave*16 + l16;
    v16bf bWh[4], bWl2[4];
    #pragma unroll
    for (int ks = 0; ks < 4; ++ks) {
        const int kb = ks*32 + lh*8;
        bWh[ks]  = ld_frag(Whi + eW*DD + kb, Whi + eW*DD + kb + 16);
        bWl2[ks] = ld_frag(Wlo + eW*DD + kb, Wlo + eW*DD + kb + 16);
    }
    const float wwE = vWw[eW];
    const float wlE = vWl[eW];

    // ---- Node loop: async double-buffered x_nb stream + fused attention
    for (int ln = 0; ln < NPB; ++ln) {
        long node = node0 + ln; if (node >= Ntot) node = Ntot - 1;
        float* stgCur = (ln & 1) ? stg1 : stg0;
        float* stgNxt = (ln & 1) ? stg0 : stg1;

        // wait for this node's async DMA (own wave), then make it block-visible
        asm volatile("s_wait_asynccnt 0x0" ::: "memory");
        __syncthreads();

        // convert staged f32 -> split bf16 operand buffers
        if (tid < KNB) sco[tid] = 0.0f;
        #pragma unroll
        for (int i = tid; i < (KNB*DD)/4; i += NTHREADS) {
            float4 f4 = ((const float4*)stgCur)[i];
            int b = i*4;
            __bf16 h0=(__bf16)f4.x; nbh[b+0]=h0; nbl[b+0]=(__bf16)(f4.x-(float)h0);
            __bf16 h1=(__bf16)f4.y; nbh[b+1]=h1; nbl[b+1]=(__bf16)(f4.y-(float)h1);
            __bf16 h2=(__bf16)f4.z; nbh[b+2]=h2; nbl[b+2]=(__bf16)(f4.z-(float)h2);
            __bf16 h3=(__bf16)f4.w; nbh[b+3]=h3; nbl[b+3]=(__bf16)(f4.w-(float)h3);
        }
        // launch next node's DMA into the other buffer (consumed next iter)
        if (ln + 1 < NPB) {
            long nn = node0 + ln + 1; if (nn >= Ntot) nn = Ntot - 1;
            async_tile_16k(x_nb + nn * (long)(KNB*DD), stgNxt, tid);
        }
        __syncthreads();

        // n_s GEMM: this wave's e-slice x both 16-row m-tiles, bf16x3 WMMA
        const float xse = xsA[ln*DD + eW];
        #pragma unroll
        for (int mt = 0; mt < 2; ++mt) {
            const int arow = mt*16 + l16;
            v8f acc = {};
            #pragma unroll
            for (int ks = 0; ks < 4; ++ks) {
                const int ka = ks*32 + lh*8;
                v16bf ahf = ld_frag(nbh + arow*DD + ka, nbh + arow*DD + ka + 16);
                v16bf alf = ld_frag(nbl + arow*DD + ka, nbl + arow*DD + ka + 16);
                acc = wmma3(ahf, alf, bWh[ks], bWl2[ks], acc);
            }
            // fused epilogue: h = n_s + x_s + w; LeakyReLU(0.1); dot with Wl
            float sc[8];
            #pragma unroll
            for (int r = 0; r < 8; ++r) {
                const int m = mt*16 + r + 8*lh;           // neighbor index
                float v = acc[r] + xse + wgt[ln*KNB + m]*wwE;
                v = (v >= 0.0f) ? v : 0.1f*v;
                sc[r] = v * wlE;
            }
            // reduce per-row partials across the 16 lanes holding that row
            #pragma unroll
            for (int off = 8; off >= 1; off >>= 1) {
                #pragma unroll
                for (int r = 0; r < 8; ++r)
                    sc[r] += __shfl_xor(sc[r], off, 16);
            }
            if (l16 == 0) {
                #pragma unroll
                for (int r = 0; r < 8; ++r)
                    atomicAdd(sco + (mt*16 + r + 8*lh), sc[r]);
            }
        }
        __syncthreads();

        // softmax over 32 neighbors (wave 0)
        if (tid < KNB) {
            float s = sco[tid] + bl0;
            float mx = s;
            for (int off = 16; off >= 1; off >>= 1) mx = fmaxf(mx, __shfl_xor(mx, off, 32));
            float p = __expf(s - mx);
            float sm = p;
            for (int off = 16; off >= 1; off >>= 1) sm += __shfl_xor(sm, off, 32);
            att[tid] = p / sm;
        }
        __syncthreads();

        // out[n,:] = attn @ x_nb[n]  (x_nb reconstructed as hi+lo, no HBM re-read)
        {
            const int e  = tid & (DD-1);
            const int hf = tid >> 7;
            float o = 0.0f;
            #pragma unroll
            for (int k = hf*16; k < hf*16 + 16; ++k)
                o += att[k] * ((float)nbh[k*DD + e] + (float)nbl[k*DD + e]);
            if (hf) osum[e] = o;
            __syncthreads();
            if (!hf) out[node*DD + e] = o + osum[e];
        }
        __syncthreads();
    }
}

extern "C" void kernel_launch(void* const* d_in, const int* in_sizes, int n_in,
                              void* d_out, int out_size, void* d_ws, size_t ws_size,
                              hipStream_t stream) {
    const float* x      = (const float*)d_in[0];
    const float* x_nb   = (const float*)d_in[1];
    const float* weight = (const float*)d_in[2];
    const float* Wx     = (const float*)d_in[3];
    const float* bx     = (const float*)d_in[4];
    const float* Wn     = (const float*)d_in[5];
    const float* bn     = (const float*)d_in[6];
    const float* Ww     = (const float*)d_in[7];
    const float* bw     = (const float*)d_in[8];
    const float* Wl     = (const float*)d_in[9];
    const float* bl     = (const float*)d_in[10];
    float* out = (float*)d_out;

    const int Ntot   = in_sizes[0] / DD;
    const int blocks = (Ntot + NPB - 1) / NPB;

    // >64KB dynamic LDS: opt in (idempotent, deterministic per call)
    (void)hipFuncSetAttribute((const void*)gnn_attn_kernel,
                              hipFuncAttributeMaxDynamicSharedMemorySize,
                              (int)SMEM_BYTES);

    gnn_attn_kernel<<<blocks, NTHREADS, SMEM_BYTES, stream>>>(
        x, x_nb, weight, Wx, bx, Wn, bn, Ww, bw, Wl, bl, out, Ntot);
}
